// FluxJointAttention_13537736917648
// MI455X (gfx1250) — matrix operand order repaired and checked by reading, not compiled, not run on hardware
//
#include <hip/hip_runtime.h>

#define SA_N 2048
#define SB_N 512
#define S_TOT 2560
#define HH 24
#define DD 128
#define DIM 3072
#define NQKV 9216

typedef __attribute__((ext_vector_type(16))) __bf16        v16bf;
typedef __attribute__((ext_vector_type(8)))  float         v8f;
typedef __attribute__((ext_vector_type(8)))  unsigned short us8;

union BF16Frag { v16bf v; us8 h[2]; };

// --- gfx1250 async global->LDS via inline asm ------------------------------
// global_load_async_to_lds_b128 vdst(lds byte addr), vaddr(64b global), off
// LDS byte address = low 32 bits of the generic pointer (LDS aperture
// truncates to addr[31:0]). Tracked by ASYNCcnt.
#define ASYNC_LDS_B128(ldsptr, gptr, OFF)                                   \
  asm volatile("global_load_async_to_lds_b128 %0, %1, off offset:" #OFF     \
               ::"v"((unsigned)(unsigned long long)(ldsptr)),               \
                 "v"((unsigned long long)(gptr))                            \
               : "memory")

__device__ __forceinline__ void wait_asynccnt0() {
#if defined(__has_builtin)
#if __has_builtin(__builtin_amdgcn_s_wait_asynccnt)
  __builtin_amdgcn_s_wait_asynccnt(0);
#else
  asm volatile("s_wait_asynccnt 0x0" ::: "memory");
#endif
#else
  asm volatile("s_wait_asynccnt 0x0" ::: "memory");
#endif
}

__device__ __forceinline__ unsigned short f2bf(float f) {
  unsigned int u = __builtin_bit_cast(unsigned int, f);
  u = (u + 0x7FFFu + ((u >> 16) & 1u)) >> 16;
  return (unsigned short)u;
}
__device__ __forceinline__ unsigned int f2bf2(float lo, float hi) {
  return (unsigned int)f2bf(lo) | ((unsigned int)f2bf(hi) << 16);
}

// ---------------------------------------------------------------------------
// Tiled GEMM with bias: C[M][N] = A[M][K] * B[K][N] + bias[N]
// A: f32 or bf16(u16) global; B: f32 global (converted to bf16 in LDS stage,
// packed pairwise -> ds_store_b32). Block tile 128x128, K-step 32,
// 256 threads = 8 waves, each wave owns a 64x32 C region (4x2 WMMA tiles).
// M%128==0, N%128==0, K%32==0 guaranteed by caller.
// ---------------------------------------------------------------------------
template <bool A_IS_BF16>
__global__ __launch_bounds__(256) void gemm_bias(
    const void* __restrict__ Aptr, const float* __restrict__ B,
    const float* __restrict__ bias, float* __restrict__ C,
    int M, int N, int K) {
  __shared__ unsigned short As[128][32];
  __shared__ unsigned short Bs[32][128];

  const int tid  = threadIdx.x;
  const int wave = tid >> 5;
  const int lane = tid & 31;
  const int lh   = lane >> 4;   // lane half
  const int lr   = lane & 15;
  const int wm   = wave >> 2;   // 0..1
  const int wn   = wave & 3;    // 0..3

  const int bm0 = blockIdx.y * 128;
  const int bn0 = blockIdx.x * 128;

  // loader mapping
  const int arow = tid >> 1;          // 0..127
  const int acol = (tid & 1) * 16;    // 0 / 16
  const int brow = tid >> 3;          // 0..31
  const int bcol = (tid & 7) * 16;    // 0..112

  v8f acc[4][2];
#pragma unroll
  for (int mi = 0; mi < 4; ++mi)
#pragma unroll
    for (int ni = 0; ni < 2; ++ni)
#pragma unroll
      for (int v = 0; v < 8; ++v) acc[mi][ni][v] = 0.0f;

  for (int k0 = 0; k0 < K; k0 += 32) {
    // --- stage A ---
    if (A_IS_BF16) {
      const unsigned short* Ab = (const unsigned short*)Aptr;
      const unsigned short* p = Ab + (size_t)(bm0 + arow) * K + k0 + acol;
      ASYNC_LDS_B128(&As[arow][acol], p, 0);
      ASYNC_LDS_B128(&As[arow][acol], p, 16);
    } else {
      const float* Af = (const float*)Aptr;
      const float* p = Af + (size_t)(bm0 + arow) * K + k0 + acol;
      unsigned int* dst = (unsigned int*)&As[arow][acol];
#pragma unroll
      for (int i = 0; i < 8; ++i) dst[i] = f2bf2(p[2 * i], p[2 * i + 1]);
      if (k0 + 32 < K)
        __builtin_prefetch(p + 32, 0, 1);  // next k-step A tile -> GL2
    }
    // --- stage B (f32 -> bf16, packed stores) ---
    {
      const float* p = B + (size_t)(k0 + brow) * N + bn0 + bcol;
      unsigned int* dst = (unsigned int*)&Bs[brow][bcol];
#pragma unroll
      for (int i = 0; i < 8; ++i) dst[i] = f2bf2(p[2 * i], p[2 * i + 1]);
      if (k0 + 32 < K)
        __builtin_prefetch(p + (size_t)32 * N, 0, 1);  // next B tile -> GL2
    }
    if (A_IS_BF16) wait_asynccnt0();
    __syncthreads();

    // A fragments: lane holds row M=lr; K chunks lh*8 and lh*8+16
    BF16Frag af[4];
#pragma unroll
    for (int mi = 0; mi < 4; ++mi) {
      const unsigned short* p = &As[wm * 64 + mi * 16 + lr][lh * 8];
      af[mi].h[0] = *(const us8*)p;
      af[mi].h[1] = *(const us8*)(p + 16);
    }
    // B fragments: lane holds column N=lr; K chunks lh*8 and lh*8+16
    BF16Frag bf[2];
#pragma unroll
    for (int ni = 0; ni < 2; ++ni) {
      const int bc = wn * 32 + ni * 16 + lr;
#pragma unroll
      for (int i = 0; i < 8; ++i) {
        bf[ni].h[0][i] = Bs[lh * 8 + i][bc];
        bf[ni].h[1][i] = Bs[lh * 8 + 16 + i][bc];
      }
    }
#pragma unroll
    for (int mi = 0; mi < 4; ++mi)
#pragma unroll
      for (int ni = 0; ni < 2; ++ni)
        acc[mi][ni] = __builtin_amdgcn_wmma_f32_16x16x32_bf16(
            false, af[mi].v, false, bf[ni].v, (short)0, acc[mi][ni], false, false);
    __syncthreads();
  }

  // epilogue: C element (lane, vgpr v) -> M = v + 8*lh, N = lr
#pragma unroll
  for (int mi = 0; mi < 4; ++mi)
#pragma unroll
    for (int ni = 0; ni < 2; ++ni) {
      const int col = bn0 + wn * 32 + ni * 16 + lr;
      const float bv = bias[col];
#pragma unroll
      for (int v = 0; v < 8; ++v) {
        const int row = bm0 + wm * 64 + mi * 16 + v + 8 * lh;
        C[(size_t)row * N + col] = acc[mi][ni][v] + bv;
      }
    }
}

// ---------------------------------------------------------------------------
// QKV post-processing: per (token s, head h): RMS-norm q/k, RoPE q/k,
// fold 1/sqrt(D) into q, emit head-major bf16 Q/K/V  [h][s][d].
// Token order for attention: [b tokens 0..511, a tokens 512..2559].
// ---------------------------------------------------------------------------
__global__ __launch_bounds__(128) void qkv_post(
    const float* __restrict__ qkvA, const float* __restrict__ qkvB,
    const float* __restrict__ rope,
    const float* __restrict__ wqa, const float* __restrict__ wka,
    const float* __restrict__ wqb, const float* __restrict__ wkb,
    unsigned short* __restrict__ Qh, unsigned short* __restrict__ Kh,
    unsigned short* __restrict__ Vh) {
  const int s = blockIdx.x;   // 0..2559 (concat order: b then a)
  const int h = blockIdx.y;   // 0..23
  const int d = threadIdx.x;  // 0..127

  const bool isB = (s < SB_N);
  const float* row = isB ? (qkvB + (size_t)s * NQKV)
                         : (qkvA + (size_t)(s - SB_N) * NQKV);
  const float* wq = isB ? wqb : wqa;
  const float* wk = isB ? wkb : wka;

  const float q = row[h * DD + d];
  const float k = row[DIM + h * DD + d];
  const float v = row[2 * DIM + h * DD + d];

  __shared__ float rq[128];
  __shared__ float rk[128];
  rq[d] = q * q;
  rk[d] = k * k;
  __syncthreads();
#pragma unroll
  for (int off = 64; off > 0; off >>= 1) {
    if (d < off) { rq[d] += rq[d + off]; rk[d] += rk[d + off]; }
    __syncthreads();
  }
  const float vq = rq[0] * (1.0f / DD);
  const float vk = rk[0] * (1.0f / DD);

  float qn = q * rsqrtf(vq + 1e-6f) * wq[d];
  float kn = k * rsqrtf(vk + 1e-6f) * wk[d];

  // RoPE: out[2i+j] = f[s,i,j,0]*x[2i] + f[s,i,j,1]*x[2i+1]
  const int i = d >> 1, j = d & 1;
  const float qp = __shfl_xor(qn, 1, 32);  // partner d^1 (same wave)
  const float kp = __shfl_xor(kn, 1, 32);
  const float qx0 = (j == 0) ? qn : qp;
  const float qx1 = (j == 0) ? qp : qn;
  const float kx0 = (j == 0) ? kn : kp;
  const float kx1 = (j == 0) ? kp : kn;
  const float* fr = rope + ((size_t)s * (DD / 2) + i) * 4 + j * 2;
  float qro = fr[0] * qx0 + fr[1] * qx1;
  float kro = fr[0] * kx0 + fr[1] * kx1;
  qro *= 0.08838834764831845f;  // 1/sqrt(128) folded into Q

  const size_t base = ((size_t)h * S_TOT + s) * DD + d;
  Qh[base] = f2bf(qro);
  Kh[base] = f2bf(kro);
  Vh[base] = f2bf(v);
}

// ---------------------------------------------------------------------------
// Flash attention: grid (S/64, H), 128 threads = 4 waves.
// Wave w owns 16 queries. Keys tiled by 32; K staged transposed in LDS so
// score-WMMA B fragments are contiguous; V tile staged with async
// global->LDS (ASYNCcnt). Online softmax in f32.
// Output bf16, token-major [s][h*128+d].
// ---------------------------------------------------------------------------
__global__ __launch_bounds__(128) void flash_attn(
    const unsigned short* __restrict__ Qh, const unsigned short* __restrict__ Kh,
    const unsigned short* __restrict__ Vh, unsigned short* __restrict__ Out) {
  __shared__ unsigned short Kt[128][32];     // [d][key] transposed
  __shared__ unsigned short Vs[32][128];     // [key][d]
  __shared__ unsigned short Ps[4][16][32];   // per-wave P tile

  const int qt = blockIdx.x, h = blockIdx.y;
  const int tid = threadIdx.x;
  const int w = tid >> 5, lane = tid & 31;
  const int lh = lane >> 4, lr = lane & 15;

  const unsigned short* Qb = Qh + (size_t)h * S_TOT * DD;
  const unsigned short* Kb = Kh + (size_t)h * S_TOT * DD;
  const unsigned short* Vb = Vh + (size_t)h * S_TOT * DD;

  const int qrow0 = qt * 64 + w * 16;

  // preload Q fragments (16 queries x 128 d = 4 A-fragments)
  BF16Frag qf[4];
#pragma unroll
  for (int ds = 0; ds < 4; ++ds) {
    const unsigned short* p = Qb + (size_t)(qrow0 + lr) * DD + ds * 32 + lh * 8;
    qf[ds].h[0] = *(const us8*)p;
    qf[ds].h[1] = *(const us8*)(p + 16);
  }

  v8f o[8];
#pragma unroll
  for (int f = 0; f < 8; ++f)
#pragma unroll
    for (int v = 0; v < 8; ++v) o[f][v] = 0.0f;
  float m[8], l[8];
#pragma unroll
  for (int v = 0; v < 8; ++v) { m[v] = -1e30f; l[v] = 0.0f; }

  const int key = tid >> 2;           // 0..31
  const int db  = (tid & 3) * 32;     // 0..96

  for (int kt = 0; kt < S_TOT / 32; ++kt) {
    const int kb = kt * 32;
    // stage V via async global->LDS; K transposed via VGPRs
    {
      const unsigned short* kp = Kb + (size_t)(kb + key) * DD + db;
      const unsigned short* vp = Vb + (size_t)(kb + key) * DD + db;
      ASYNC_LDS_B128(&Vs[key][db], vp, 0);
      ASYNC_LDS_B128(&Vs[key][db], vp, 16);
      ASYNC_LDS_B128(&Vs[key][db], vp, 32);
      ASYNC_LDS_B128(&Vs[key][db], vp, 48);
      us8 t0 = *(const us8*)kp;
      us8 t1 = *(const us8*)(kp + 8);
      us8 t2 = *(const us8*)(kp + 16);
      us8 t3 = *(const us8*)(kp + 24);
#pragma unroll
      for (int i = 0; i < 8; ++i) {
        Kt[db + i][key]      = t0[i];
        Kt[db + 8 + i][key]  = t1[i];
        Kt[db + 16 + i][key] = t2[i];
        Kt[db + 24 + i][key] = t3[i];
      }
      wait_asynccnt0();
    }
    __syncthreads();

    // scores: 16 queries x 32 keys = two 16x16 C tiles, K-loop over d
    v8f c0, c1;
#pragma unroll
    for (int v = 0; v < 8; ++v) { c0[v] = 0.0f; c1[v] = 0.0f; }
#pragma unroll
    for (int ds = 0; ds < 4; ++ds) {
      BF16Frag b0, b1;
#pragma unroll
      for (int i = 0; i < 8; ++i) {
        b0.h[0][i] = Kt[ds * 32 + lh * 8 + i][lr];
        b0.h[1][i] = Kt[ds * 32 + lh * 8 + 16 + i][lr];
        b1.h[0][i] = Kt[ds * 32 + lh * 8 + i][16 + lr];
        b1.h[1][i] = Kt[ds * 32 + lh * 8 + 16 + i][16 + lr];
      }
      c0 = __builtin_amdgcn_wmma_f32_16x16x32_bf16(false, qf[ds].v, false, b0.v,
                                                   (short)0, c0, false, false);
      c1 = __builtin_amdgcn_wmma_f32_16x16x32_bf16(false, qf[ds].v, false, b1.v,
                                                   (short)0, c1, false, false);
    }

    // online softmax (row M = v + 8*lh lives on a 16-lane group)
    float corr[8];
#pragma unroll
    for (int v = 0; v < 8; ++v) {
      float x0 = c0[v], x1 = c1[v];
      float mx = fmaxf(x0, x1);
#pragma unroll
      for (int off = 8; off > 0; off >>= 1) mx = fmaxf(mx, __shfl_xor(mx, off, 16));
      const float mn = fmaxf(m[v], mx);
      corr[v] = __expf(m[v] - mn);
      m[v] = mn;
      const float p0 = __expf(x0 - mn);
      const float p1 = __expf(x1 - mn);
      float rs = p0 + p1;
#pragma unroll
      for (int off = 8; off > 0; off >>= 1) rs += __shfl_xor(rs, off, 16);
      l[v] = l[v] * corr[v] + rs;
      const int row = v + 8 * lh;
      Ps[w][row][lr]      = f2bf(p0);
      Ps[w][row][16 + lr] = f2bf(p1);
    }
#pragma unroll
    for (int f = 0; f < 8; ++f)
#pragma unroll
      for (int v = 0; v < 8; ++v) o[f][v] *= corr[v];
    __syncthreads();

    // P @ V : A = P (16x32 keys), B = V (32 keys x 16 d) per d-tile
    BF16Frag pa;
    {
      const unsigned short* p = &Ps[w][lr][lh * 8];
      pa.h[0] = *(const us8*)p;
      pa.h[1] = *(const us8*)(p + 16);
    }
#pragma unroll
    for (int f = 0; f < 8; ++f) {
      BF16Frag bv;
#pragma unroll
      for (int i = 0; i < 8; ++i) {
        bv.h[0][i] = Vs[lh * 8 + i][f * 16 + lr];
        bv.h[1][i] = Vs[lh * 8 + 16 + i][f * 16 + lr];
      }
      o[f] = __builtin_amdgcn_wmma_f32_16x16x32_bf16(false, pa.v, false, bv.v,
                                                     (short)0, o[f], false, false);
    }
    __syncthreads();
  }

  // normalize + write bf16 token-major [s][h*128 + d]
#pragma unroll
  for (int v = 0; v < 8; ++v) {
    const float inv = (l[v] > 0.0f) ? (1.0f / l[v]) : 0.0f;
    const int row = qrow0 + v + 8 * lh;
#pragma unroll
    for (int f = 0; f < 8; ++f) {
      const int col = h * DD + f * 16 + lr;
      Out[(size_t)row * DIM + col] = f2bf(o[f][v] * inv);
    }
  }
}

// ---------------------------------------------------------------------------
extern "C" void kernel_launch(void* const* d_in, const int* in_sizes, int n_in,
                              void* d_out, int out_size, void* d_ws, size_t ws_size,
                              hipStream_t stream) {
  (void)in_sizes; (void)n_in; (void)out_size; (void)ws_size;
  const float* hidden_a = (const float*)d_in[0];
  const float* hidden_b = (const float*)d_in[1];
  const float* rope     = (const float*)d_in[2];
  const float* Wqkv_a   = (const float*)d_in[3];
  const float* bqkv_a   = (const float*)d_in[4];
  const float* Wqkv_b   = (const float*)d_in[5];
  const float* bqkv_b   = (const float*)d_in[6];
  const float* wq_a     = (const float*)d_in[7];
  const float* wk_a     = (const float*)d_in[8];
  const float* wq_b     = (const float*)d_in[9];
  const float* wk_b     = (const float*)d_in[10];
  const float* Wout_a   = (const float*)d_in[11];
  const float* bout_a   = (const float*)d_in[12];
  const float* Wout_b   = (const float*)d_in[13];
  const float* bout_b   = (const float*)d_in[14];
  float* out = (float*)d_out;

  char* ws = (char*)d_ws;
  float* qkvA = (float*)ws;                                  // 2048*9216 f32
  float* qkvB = (float*)(ws + 75497472);                     // 512*9216  f32
  unsigned short* Qh   = (unsigned short*)(ws + 94371840);   // 24*2560*128 bf16
  unsigned short* Kh   = (unsigned short*)(ws + 110100480);
  unsigned short* Vh   = (unsigned short*)(ws + 125829120);
  unsigned short* attn = (unsigned short*)(ws + 141557760);  // 2560*3072 bf16

  // QKV projections (f32 A, bf16 WMMA)
  gemm_bias<false><<<dim3(NQKV / 128, SA_N / 128), 256, 0, stream>>>(
      hidden_a, Wqkv_a, bqkv_a, qkvA, SA_N, NQKV, DIM);
  gemm_bias<false><<<dim3(NQKV / 128, SB_N / 128), 256, 0, stream>>>(
      hidden_b, Wqkv_b, bqkv_b, qkvB, SB_N, NQKV, DIM);

  // RMS-norm + RoPE + head-major bf16 repack
  qkv_post<<<dim3(S_TOT, HH), 128, 0, stream>>>(
      qkvA, qkvB, rope, wq_a, wk_a, wq_b, wk_b, Qh, Kh, Vh);

  // flash attention
  flash_attn<<<dim3(S_TOT / 64, HH), 128, 0, stream>>>(Qh, Kh, Vh, attn);

  // output projections (bf16 A): d_out = [out_a (2048x3072), out_b (512x3072)]
  gemm_bias<true><<<dim3(DIM / 128, SA_N / 128), 256, 0, stream>>>(
      attn + (size_t)SB_N * DIM, Wout_a, bout_a, out, SA_N, DIM, DIM);
  gemm_bias<true><<<dim3(DIM / 128, SB_N / 128), 256, 0, stream>>>(
      attn, Wout_b, bout_b, out + (size_t)SA_N * DIM, SB_N, DIM, DIM);
}
